// LSTM_69733089017902
// MI455X (gfx1250) — matrix operand order repaired and checked by reading, not compiled, not run on hardware
//
#include <hip/hip_runtime.h>
#include <hip/hip_bf16.h>

typedef _Float16 v16h __attribute__((ext_vector_type(16)));
typedef _Float16 h8   __attribute__((ext_vector_type(8)));
typedef float    v8f  __attribute__((ext_vector_type(8)));

#define Hdim 64
#define Tlen 512
#define Bsz  2048
#define HS   72    // h row stride in f16: 64 cols + pad; rows 16B-aligned, bank-conflict-free

// gfx1250 has hardware v_tanh_f32 (TRANS op, co-executes with WMMA)
#if __has_builtin(__builtin_amdgcn_tanhf)
__device__ __forceinline__ float tanh_hw(float x) { return __builtin_amdgcn_tanhf(x); }
#else
__device__ __forceinline__ float tanh_hw(float x) {
    float e = __expf(-2.f * fabsf(x));
    float t = (1.f - e) * __builtin_amdgcn_rcpf(1.f + e);
    return copysignf(t, x);
}
#endif
__device__ __forceinline__ float sigf(float x) { return 0.5f * tanh_hw(0.5f * x) + 0.5f; }

// ---------------------------------------------------------------------------
// Bidirectional LSTM. One workgroup = (direction, 16 batch rows), 4 waves.
// z(16x256) = [h | x | 1 | 0](16x96 f16) @ B(96x256 f16), all via
// v_wmma_f32_16x16x32_f16. Wave w owns columns w*16..w*16+15 of ALL FOUR
// gates, so its 4 accumulators are (z_i, z_f, z_g, z_o) for the same cells:
// the LSTM cell update runs entirely on accumulator registers — no z buffer,
// one barrier per step, h double-buffered in LDS.
// ---------------------------------------------------------------------------
__global__ __launch_bounds__(128) void lstm_kernel(
    const float* __restrict__ y,
    const float* __restrict__ Wih_f, const float* __restrict__ Whh_f,
    const float* __restrict__ bih_f, const float* __restrict__ bhh_f,
    const float* __restrict__ Wih_b, const float* __restrict__ Whh_b,
    const float* __restrict__ bih_b, const float* __restrict__ bhh_b,
    float* __restrict__ h_out)        // [2][Bsz][Hdim]
{
    __shared__ __align__(16) _Float16 hbuf[2][16 * HS];   // double-buffered h (f16)

    const int tid  = threadIdx.x;
    const int lane = tid & 31;
    const int wave = tid >> 5;
    const int dir  = blockIdx.x >> 7;   // 0 = forward, 1 = backward
    const int tile = blockIdx.x & 127;

    const float* Wih = dir ? Wih_b : Wih_f;
    const float* Whh = dir ? Whh_b : Whh_f;
    const float* bih = dir ? bih_b : bih_f;
    const float* bhh = dir ? bhh_b : bhh_f;

    const int nloc  = lane & 15;        // column within 16-wide tile / A row
    const int hsel  = lane >> 4;        // lane half
    const int khalf = hsel * 16;        // B-layout K sub-half

    // ---- B fragments (loop-invariant registers) ---------------------------
    // Bf[g][kc] : gate g, K-chunk kc.  Global gate index n = g*64 + wave*16 + nloc.
    // B layout used here (contiguous K, same order as A): lanes 0-15 hold K kb..kb+15, lanes 16-31 the next 16;
    // VGPR v holds K = kb+2v (lo16), kb+2v+1 (hi16).
    v16h Bf[4][3];
#pragma unroll
    for (int g = 0; g < 4; ++g) {
        const int n = g * 64 + wave * 16 + nloc;
#pragma unroll
        for (int kc = 0; kc < 3; ++kc) {
            const int kb = kc * 32 + khalf;
            v16h bf;
#pragma unroll
            for (int i = 0; i < 16; ++i) {
                const int k = kb + i;
                float v;
                if (k < 64)       v = Whh[n * 64 + k];        // W_hh^T
                else if (k < 68)  v = Wih[n * 4 + (k - 64)];  // W_ih^T
                else if (k == 68) v = bih[n] + bhh[n];        // bias row
                else              v = 0.f;                    // K padding
                bf[i] = (_Float16)v;
            }
            Bf[g][kc] = bf;
        }
    }

    // ---- init LDS h buffers to zero ---------------------------------------
    {
        _Float16* hp = &hbuf[0][0];
        for (int i = tid; i < 2 * 16 * HS; i += 128) hp[i] = (_Float16)0.f;
    }

    // cell state in registers: lane owns rows hsel*8..hsel*8+7, col wave*16+nloc
    float creg[8];
#pragma unroll
    for (int r = 0; r < 8; ++r) creg[r] = 0.f;

    // A-fragment addressing (contiguous K, same order as B):
    // lanes 0-15: elems 0-15 = K b0..b0+15 (b0 = kc*32)
    // lanes 16-31: same with +16 offset.
    const int ab0   = hsel * 16;
    const float xmask = (hsel == 0) ? 1.f : 0.f;   // x|1 live only in lane half 0

    // software-pipelined x load: one float4 of y per lane (row = nloc)
    const float4* yv   = (const float4*)y;
    const size_t  brow = (size_t)(tile * 16 + nloc) * Tlen;
    float4 xreg = yv[brow + (dir ? (Tlen - 1) : 0)];

    for (int s = 0; s < Tlen; ++s) {
        // a2 = [x0 x1 x2 x3 1 0...] for lane half 0, zeros for half 1 (registers only)
        v16h a2 = {};
        a2[0] = (_Float16)(xreg.x * xmask);
        a2[1] = (_Float16)(xreg.y * xmask);
        a2[2] = (_Float16)(xreg.z * xmask);
        a2[3] = (_Float16)(xreg.w * xmask);
        a2[4] = (_Float16)xmask;

        // prefetch next step's x (hides L2 latency behind this step's compute)
        const int sn = (s + 1 < Tlen) ? s + 1 : s;
        float4 xnext = yv[brow + (dir ? (Tlen - 1 - sn) : sn)];

        __syncthreads();   // h(s) visible in hbuf[s&1]

        // A fragments for the h part (two ds_read_b128 each)
        v16h a0, a1;
        {
            const _Float16* hrow = &hbuf[s & 1][nloc * HS];
            h8 lo0 = *(const h8*)(hrow + ab0);
            h8 hi0 = *(const h8*)(hrow + ab0 + 8);
            h8 lo1 = *(const h8*)(hrow + 32 + ab0);
            h8 hi1 = *(const h8*)(hrow + 32 + ab0 + 8);
#pragma unroll
            for (int i = 0; i < 8; ++i) {
                a0[i] = lo0[i]; a0[i + 8] = hi0[i];
                a1[i] = lo1[i]; a1[i + 8] = hi1[i];
            }
        }

        // 12 WMMAs: 4 independent gate chains of 3
        v8f acc[4];
#pragma unroll
        for (int g = 0; g < 4; ++g) {
            v8f a = {};
            a = __builtin_amdgcn_wmma_f32_16x16x32_f16(false, a0, false, Bf[g][0],
                                                       (short)0, a, false, false);
            a = __builtin_amdgcn_wmma_f32_16x16x32_f16(false, a1, false, Bf[g][1],
                                                       (short)0, a, false, false);
            a = __builtin_amdgcn_wmma_f32_16x16x32_f16(false, a2, false, Bf[g][2],
                                                       (short)0, a, false, false);
            acc[g] = a;
        }

        // LSTM cell update, fully in accumulator registers.
        // acc layout: VGPR r, this lane -> row M = hsel*8 + r, col j = wave*16 + nloc
#pragma unroll
        for (int r = 0; r < 8; ++r) {
            const float zi = acc[0][r];
            const float zf = acc[1][r];
            const float zg = acc[2][r];
            const float zo = acc[3][r];
            const float cn = sigf(zf) * creg[r] + sigf(zi) * tanh_hw(zg);
            creg[r] = cn;
            const float hn = sigf(zo) * tanh_hw(cn);
            if (s == Tlen - 1) {
                h_out[((size_t)dir * Bsz + tile * 16 + hsel * 8 + r) * Hdim
                      + wave * 16 + nloc] = hn;
            } else {
                hbuf[(s + 1) & 1][(hsel * 8 + r) * HS + wave * 16 + nloc] =
                    (_Float16)hn;
            }
        }

        xreg = xnext;
    }
}

// ---------------------------------------------------------------------------
// Weight-norm scales: scale_i = g_i / ||v_i||_F   (3 blocks)
// ---------------------------------------------------------------------------
__global__ void norm_kernel(const float* __restrict__ v1,
                            const float* __restrict__ v2,
                            const float* __restrict__ v3,
                            const float* __restrict__ g1,
                            const float* __restrict__ g2,
                            const float* __restrict__ g3,
                            float* __restrict__ scales)
{
    __shared__ float red[256];
    const int which = blockIdx.x;
    const float* v = (which == 0) ? v1 : (which == 1) ? v2 : v3;
    const float* g = (which == 0) ? g1 : (which == 1) ? g2 : g3;
    const int n = (which == 0) ? 256 * 128 : (which == 1) ? 128 * 256 : 4 * 128;

    float s = 0.f;
    for (int i = threadIdx.x; i < n; i += 256) { float x = v[i]; s += x * x; }
    red[threadIdx.x] = s;
    __syncthreads();
    for (int off = 128; off > 0; off >>= 1) {
        if (threadIdx.x < off) red[threadIdx.x] += red[threadIdx.x + off];
        __syncthreads();
    }
    if (threadIdx.x == 0) scales[which] = g[0] / sqrtf(red[0]);
}

// ---------------------------------------------------------------------------
// MLP head: one block per batch row; 128 -> 256 -> 128 -> 4 -> sigmoid.
// ~0.3 GFLOP total; weights L2-resident.
// ---------------------------------------------------------------------------
__global__ __launch_bounds__(128) void head_kernel(
    const float* __restrict__ hbuf,      // [2][Bsz][64]
    const float* __restrict__ v1, const float* __restrict__ b1,
    const float* __restrict__ v2, const float* __restrict__ b2,
    const float* __restrict__ v3, const float* __restrict__ b3,
    const float* __restrict__ scales,
    float* __restrict__ out)
{
    __shared__ float hc[128];
    __shared__ float x1[256];
    __shared__ float x2[128];
    const int t = threadIdx.x;
    const int b = blockIdx.x;

    hc[t] = (t < 64) ? hbuf[(size_t)b * 64 + t]
                     : hbuf[(size_t)Bsz * 64 + (size_t)b * 64 + (t - 64)];
    __syncthreads();

    const float s1 = scales[0], s2 = scales[1], s3 = scales[2];

    for (int n = t; n < 256; n += 128) {
        const float* w = &v1[n * 128];
        float acc = 0.f;
#pragma unroll 4
        for (int k = 0; k < 128; ++k) acc += w[k] * hc[k];
        x1[n] = s1 * acc + b1[n];
    }
    __syncthreads();
    {
        const float* w = &v2[t * 256];
        float acc = 0.f;
#pragma unroll 4
        for (int k = 0; k < 256; ++k) acc += w[k] * x1[k];
        x2[t] = s2 * acc + b2[t];
    }
    __syncthreads();
    if (t < 4) {
        const float* w = &v3[t * 128];
        float acc = 0.f;
#pragma unroll 4
        for (int k = 0; k < 128; ++k) acc += w[k] * x2[k];
        out[(size_t)b * 4 + t] = 0.5f * tanh_hw(0.5f * (s3 * acc + b3[t])) + 0.5f;
    }
}

// ---------------------------------------------------------------------------
extern "C" void kernel_launch(void* const* d_in, const int* in_sizes, int n_in,
                              void* d_out, int out_size, void* d_ws, size_t ws_size,
                              hipStream_t stream)
{
    (void)in_sizes; (void)n_in; (void)out_size; (void)ws_size;

    const float* y     = (const float*)d_in[0];
    const float* Wih_f = (const float*)d_in[1];
    const float* Whh_f = (const float*)d_in[2];
    const float* bih_f = (const float*)d_in[3];
    const float* bhh_f = (const float*)d_in[4];
    const float* Wih_b = (const float*)d_in[5];
    const float* Whh_b = (const float*)d_in[6];
    const float* bih_b = (const float*)d_in[7];
    const float* bhh_b = (const float*)d_in[8];
    const float* v1 = (const float*)d_in[9];
    const float* g1 = (const float*)d_in[10];
    const float* b1 = (const float*)d_in[11];
    const float* v2 = (const float*)d_in[12];
    const float* g2 = (const float*)d_in[13];
    const float* b2 = (const float*)d_in[14];
    const float* v3 = (const float*)d_in[15];
    const float* g3 = (const float*)d_in[16];
    const float* b3 = (const float*)d_in[17];

    float* ws     = (float*)d_ws;
    float* h_all  = ws;                       // 2 * 2048 * 64 floats
    float* scales = ws + 2 * Bsz * Hdim;      // 3 floats

    lstm_kernel<<<256, 128, 0, stream>>>(y, Wih_f, Whh_f, bih_f, bhh_f,
                                         Wih_b, Whh_b, bih_b, bhh_b, h_all);
    norm_kernel<<<3, 256, 0, stream>>>(v1, v2, v3, g1, g2, g3, scales);
    head_kernel<<<2048, 128, 0, stream>>>(h_all, v1, b1, v2, b2, v3, b3,
                                          scales, (float*)d_out);
}
